// Mamba2TemporalCore_1408749273203
// MI455X (gfx1250) — compile-verified, matrix operand order checked by reading, not run on hardware
//
#include <hip/hip_runtime.h>
#include <hip/hip_bf16.h>

// ---- shapes (match reference) ----
#define DM      128      // D_MODEL
#define EDIM    256      // EXPAND*D_MODEL
#define NSTATE  64       // D_STATE
#define DTRANK  8
#define TSEQ    1024
#define BSZ     4
#define MROWS   (BSZ*TSEQ)       // 4096
#define N1      (2*EDIM)         // 512  (in_proj out)
#define N2      (2*NSTATE+EDIM)  // 384  (B,C,delta_pre)
#define EPSF    1e-5f

typedef __attribute__((ext_vector_type(16))) _Float16 v16h;
typedef __attribute__((ext_vector_type(8)))  _Float16 v8h;
typedef __attribute__((ext_vector_type(8)))  float    v8f;

// Load a 16x32 f16 A/B WMMA fragment slice for this lane.
// p points at row base + half*8 (elements). lo = K 0..7(+half*8), hi = +16.
static __device__ __forceinline__ v16h load_frag16(const _Float16* p) {
    union { v16h v; v8h h[2]; } u;
    u.h[0] = *(const v8h*)(p);
    u.h[1] = *(const v8h*)(p + 16);
    return u.v;
}

// ---------------- WMMA GEMM: C[M,N] = A[M,K] * Bw[N,K]^T ----------------
// A row-major MxK (f16), Bw row-major NxK (f16) => fragments k-contiguous.
// Register-blocked: each wave computes a 16(M) x 64(N) strip: one A fragment
// per K-step feeds 4 WMMAs. block(32,4) = 4 waves = 64x64 C-tile per block.
// Requires M % 64 == 0, N % 64 == 0, K % 32 == 0.
__global__ void gemm_wmma_f16(const _Float16* __restrict__ A,
                              const _Float16* __restrict__ Bw,
                              float* __restrict__ C,
                              int M, int N, int K) {
    const int lane = threadIdx.x;          // 0..31
    const int mr   = lane & 15;
    const int half = lane >> 4;
    const int m0   = (blockIdx.y * 4 + threadIdx.y) * 16;
    const int n0   = blockIdx.x * 64;

    const _Float16* ap = A + (size_t)(m0 + mr) * K + half * 8;
    const _Float16* bp = Bw + (size_t)(n0 + mr) * K + half * 8;
    const size_t bstride = (size_t)16 * K;   // 16 N-rows of Bw

    v8f acc[4] = {{}, {}, {}, {}};
    for (int k0 = 0; k0 < K; k0 += 32) {
        v16h a = load_frag16(ap + k0);
#pragma unroll
        for (int j = 0; j < 4; ++j) {
            v16h b = load_frag16(bp + j * bstride + k0);
            acc[j] = __builtin_amdgcn_wmma_f32_16x16x32_f16(
                /*neg_a=*/false, a, /*neg_b=*/false, b,
                /*c_mod=*/(short)0, acc[j], /*reuse_a=*/false, /*reuse_b=*/false);
        }
    }
    // D layout: VGPR r -> row m0 + 8*half + r, col n0 + 16*j + mr
#pragma unroll
    for (int j = 0; j < 4; ++j) {
        float* cp = C + (size_t)(m0 + 8 * half) * N + n0 + 16 * j + mr;
#pragma unroll
        for (int r = 0; r < 8; ++r) cp[(size_t)r * N] = acc[j][r];
    }
}

// ---------------- small helpers ----------------
__global__ void f32_to_f16_kernel(const float* __restrict__ s,
                                  _Float16* __restrict__ d, int n) {
    int i = blockIdx.x * 256 + threadIdx.x;
    if (i < n) d[i] = (_Float16)s[i];
}

// Wcat[N2=384][256]: rows 0..127 = x_proj_W[8..135]; rows 128..383 = dt_W @ x_proj_W[:8]
__global__ void build_wcat_kernel(const float* __restrict__ x_proj_W,
                                  const float* __restrict__ dt_W,
                                  _Float16* __restrict__ wcat) {
    int i = blockIdx.x * 256 + threadIdx.x;   // grid 384 * 256
    int n = i >> 8, k = i & 255;
    float v;
    if (n < 2 * NSTATE) {
        v = x_proj_W[(size_t)(DTRANK + n) * EDIM + k];
    } else {
        int e = n - 2 * NSTATE;
        v = 0.f;
#pragma unroll
        for (int r = 0; r < DTRANK; ++r)
            v += dt_W[e * DTRANK + r] * x_proj_W[(size_t)r * EDIM + k];
    }
    wcat[i] = (_Float16)v;
}

static __device__ __forceinline__ float block_sum_128(float v, float* sbuf) {
#pragma unroll
    for (int off = 16; off > 0; off >>= 1) v += __shfl_xor(v, off, 32);
    int wid = threadIdx.x >> 5;
    if ((threadIdx.x & 31) == 0) sbuf[wid] = v;
    __syncthreads();
    float r = sbuf[0] + sbuf[1] + sbuf[2] + sbuf[3];
    __syncthreads();
    return r;
}

// z = z_seq + aux@aux_W^T + aux_b ; zn = RMSNorm(z)*rms_w  (f16 for GEMM1 A)
__global__ void prelude_kernel(const float* __restrict__ z_seq,
                               const float* __restrict__ aux_t,
                               const float* __restrict__ aux_W,
                               const float* __restrict__ aux_b,
                               const float* __restrict__ rms_w,
                               float* __restrict__ z,
                               _Float16* __restrict__ zn_h) {
    __shared__ float sbuf[4];
    int row = blockIdx.x, d = threadIdx.x;      // block 128
    float a0 = aux_t[row * 3 + 0], a1 = aux_t[row * 3 + 1], a2 = aux_t[row * 3 + 2];
    float v = z_seq[(size_t)row * DM + d]
            + a0 * aux_W[d * 3 + 0] + a1 * aux_W[d * 3 + 1] + a2 * aux_W[d * 3 + 2]
            + aux_b[d];
    z[(size_t)row * DM + d] = v;
    float ss = block_sum_128(v * v, sbuf);
    float inv = rsqrtf(ss * (1.f / DM) + EPSF);
    zn_h[(size_t)row * DM + d] = (_Float16)(v * inv * rms_w[d]);
}

// causal depthwise conv k=4 over time on xz[:, :EDIM], + bias, silu
__global__ void conv_silu_kernel(const float* __restrict__ xz,
                                 const float* __restrict__ conv_W,
                                 const float* __restrict__ conv_b,
                                 float* __restrict__ xs,
                                 _Float16* __restrict__ xs_h) {
    int row = blockIdx.x, e = threadIdx.x;      // block 256
    int t = row & (TSEQ - 1);
    float acc = conv_b[e];
#pragma unroll
    for (int j = 0; j < 4; ++j) {
        int tt = t - 3 + j;
        if (tt >= 0)
            acc += conv_W[e * 4 + j] * xz[(size_t)(row - 3 + j) * N1 + e];
    }
    float s = acc / (1.f + expf(-acc));         // silu
    xs[(size_t)row * EDIM + e] = s;
    xs_h[(size_t)row * EDIM + e] = (_Float16)s;
}

// selective scan: one wave32 per (b,e); lane owns states n=lane and n=lane+32
__global__ void scan_kernel(const float* __restrict__ dbc,   // [M, 384] B|C|delta_pre
                            const float* __restrict__ xs,    // [M, 256]
                            const float* __restrict__ xz,    // [M, 512] (zg in cols 256..)
                            const float* __restrict__ A_log,
                            const float* __restrict__ D_param,
                            const float* __restrict__ dt_b,
                            _Float16* __restrict__ ysh) {
    int w = blockIdx.x * (blockDim.x >> 5) + (threadIdx.x >> 5);  // 0..1023
    int lane = threadIdx.x & 31;
    int b = w >> 8, e = w & 255;
    float A0 = -expf(A_log[e * NSTATE + lane]);
    float A1 = -expf(A_log[e * NSTATE + lane + 32]);
    float Dp = D_param[e], dtb = dt_b[e];
    float h0 = 0.f, h1 = 0.f;
    size_t row = (size_t)b * TSEQ;
    for (int t = 0; t < TSEQ; ++t, ++row) {
        const float* dr = dbc + row * N2;
        // overlap next timestep's row fill (L2 -> L0) with this step's math
        __builtin_prefetch(dr + N2 + lane, 0, 3);       // global_prefetch_b8
        float dpre = dr[2 * NSTATE + e] + dtb;
        float delta = (dpre > 20.f) ? dpre : log1pf(expf(dpre));
        float x  = xs[row * EDIM + e];
        float dx = delta * x;
        h0 = expf(delta * A0) * h0 + dx * dr[lane];
        h1 = expf(delta * A1) * h1 + dx * dr[lane + 32];
        float acc = h0 * dr[NSTATE + lane] + h1 * dr[NSTATE + lane + 32];
#pragma unroll
        for (int off = 16; off > 0; off >>= 1) acc += __shfl_xor(acc, off, 32);
        if (lane == 0) {
            float zg = xz[row * N1 + EDIM + e];
            float y = (acc + Dp * x) * (zg / (1.f + expf(-zg)));
            ysh[row * EDIM + e] = (_Float16)y;
        }
    }
}

// out = LayerNorm(gemm3 + 2*z) * ln_w + ln_b
__global__ void ln_kernel(const float* __restrict__ g3,
                          const float* __restrict__ z,
                          const float* __restrict__ ln_w,
                          const float* __restrict__ ln_b,
                          float* __restrict__ out) {
    __shared__ float sbuf[4];
    int row = blockIdx.x, d = threadIdx.x;      // block 128
    float v = g3[(size_t)row * DM + d] + 2.f * z[(size_t)row * DM + d];
    float mu = block_sum_128(v, sbuf) * (1.f / DM);
    float c = v - mu;
    float var = block_sum_128(c * c, sbuf) * (1.f / DM);
    out[(size_t)row * DM + d] = c * rsqrtf(var + EPSF) * ln_w[d] + ln_b[d];
}

extern "C" void kernel_launch(void* const* d_in, const int* in_sizes, int n_in,
                              void* d_out, int out_size, void* d_ws, size_t ws_size,
                              hipStream_t stream) {
    const float* z_seq      = (const float*)d_in[0];
    const float* aux_tensor = (const float*)d_in[1];
    const float* aux_W      = (const float*)d_in[2];
    const float* aux_b      = (const float*)d_in[3];
    const float* ln_w       = (const float*)d_in[4];
    const float* ln_b       = (const float*)d_in[5];
    const float* rms_w      = (const float*)d_in[6];
    const float* in_proj_W  = (const float*)d_in[7];   // (512,128)
    const float* conv_W     = (const float*)d_in[8];   // (256,1,4)
    const float* conv_b     = (const float*)d_in[9];
    const float* x_proj_W   = (const float*)d_in[10];  // (136,256)
    const float* dt_W       = (const float*)d_in[11];  // (256,8)
    const float* dt_b       = (const float*)d_in[12];
    const float* A_log      = (const float*)d_in[13];  // (256,64)
    const float* D_param    = (const float*)d_in[14];
    const float* out_proj_W = (const float*)d_in[15];  // (128,256)
    float* out = (float*)d_out;

    // ---- workspace carve (256B aligned), total ~27 MB ----
    char* ws = (char*)d_ws;
    auto carve = [&](size_t bytes) -> void* {
        void* p = (void*)ws; ws += (bytes + 255) & ~(size_t)255; return p;
    };
    float*    z     = (float*)   carve((size_t)MROWS * DM * 4);    // 2 MB
    _Float16* zn_h  = (_Float16*)carve((size_t)MROWS * DM * 2);    // 1 MB
    _Float16* w1h   = (_Float16*)carve((size_t)N1 * DM * 2);       // 128 KB
    float*    xz    = (float*)   carve((size_t)MROWS * N1 * 4);    // 8 MB
    float*    xs    = (float*)   carve((size_t)MROWS * EDIM * 4);  // 4 MB
    _Float16* xs_h  = (_Float16*)carve((size_t)MROWS * EDIM * 2);  // 2 MB
    _Float16* wcat  = (_Float16*)carve((size_t)N2 * EDIM * 2);     // 192 KB
    float*    dbc   = (float*)   carve((size_t)MROWS * N2 * 4);    // 6 MB
    _Float16* ysh   = (_Float16*)carve((size_t)MROWS * EDIM * 2);  // 2 MB
    _Float16* w3h   = (_Float16*)carve((size_t)DM * EDIM * 2);     // 64 KB
    float*    g3    = (float*)   carve((size_t)MROWS * DM * 4);    // 2 MB

    // ---- weight prep (deterministic, every call) ----
    f32_to_f16_kernel<<<(N1 * DM + 255) / 256, 256, 0, stream>>>(in_proj_W, w1h, N1 * DM);
    f32_to_f16_kernel<<<(DM * EDIM + 255) / 256, 256, 0, stream>>>(out_proj_W, w3h, DM * EDIM);
    build_wcat_kernel<<<N2, 256, 0, stream>>>(x_proj_W, dt_W, wcat);

    // ---- pipeline ----
    prelude_kernel<<<MROWS, DM, 0, stream>>>(z_seq, aux_tensor, aux_W, aux_b, rms_w, z, zn_h);

    // GEMM1: xz[4096,512] = zn[4096,128] * in_proj_W[512,128]^T
    gemm_wmma_f16<<<dim3(N1 / 64, MROWS / 64), dim3(32, 4), 0, stream>>>(
        zn_h, w1h, xz, MROWS, N1, DM);

    conv_silu_kernel<<<MROWS, EDIM, 0, stream>>>(xz, conv_W, conv_b, xs, xs_h);

    // GEMM2: dbc[4096,384] = xs[4096,256] * Wcat[384,256]^T
    gemm_wmma_f16<<<dim3(N2 / 64, MROWS / 64), dim3(32, 4), 0, stream>>>(
        xs_h, wcat, dbc, MROWS, N2, EDIM);

    // selective scan: 1024 waves, 8 waves/block
    scan_kernel<<<(BSZ * EDIM) / 8, 256, 0, stream>>>(dbc, xs, xz, A_log, D_param, dt_b, ysh);

    // GEMM3: g3[4096,128] = ys[4096,256] * out_proj_W[128,256]^T
    gemm_wmma_f16<<<dim3(DM / 64, MROWS / 64), dim3(32, 4), 0, stream>>>(
        ysh, w3h, g3, MROWS, DM, EDIM);

    ln_kernel<<<MROWS, DM, 0, stream>>>(g3, z, ln_w, ln_b, out);
}
